// Head_27986006901378
// MI455X (gfx1250) — matrix operand order repaired and checked by reading, not compiled, not run on hardware
//
#include <hip/hip_runtime.h>
#include <hip/hip_bf16.h>

// Causal single-head attention, fused flash-attention style, CDNA5 WMMA path.
//
// Shapes: x[8,2048,1024] f32, W*[1024,64] f32 -> out[8,2048,64] f32.
// Pipeline (all intermediates f16 in d_ws, L2-resident):
//   1) prep_w:  W* f32[C,H] -> f16 transposed [H,C]     (B-frags contiguous)
//   2) proj:    Q=(xWq)*H^-0.5, K=xWk as f16 [b,t,h]; V as f16 [b,h,t]
//   3) attn:    4 consecutive 16-query tiles per block share double-buffered
//               K/V LDS tiles; online softmax; S and O via wmma f16.

typedef __attribute__((ext_vector_type(16))) _Float16 v16h;
typedef __attribute__((ext_vector_type(8)))  _Float16 v8h;
typedef __attribute__((ext_vector_type(8)))  float    v8f;

static constexpr int Bd = 8;
static constexpr int Td = 2048;
static constexpr int Cd = 1024;
static constexpr int Hd = 64;

#define WMMA_F16(A, B, C) \
    __builtin_amdgcn_wmma_f32_16x16x32_f16(false, (A), false, (B), (short)0, (C), false, false)

// ---------------------------------------------------------------- prep_w ----
__global__ void prep_w_kernel(const float* __restrict__ Wk,
                              const float* __restrict__ Wq,
                              const float* __restrict__ Wv,
                              _Float16* __restrict__ WkT,
                              _Float16* __restrict__ WqT,
                              _Float16* __restrict__ WvT) {
    int i = blockIdx.x * blockDim.x + threadIdx.x;      // over C*H
    if (i < Cd * Hd) {
        int c = i >> 6;
        int h = i & 63;
        int o = h * Cd + c;                             // [H][C] transposed
        WkT[o] = (_Float16)Wk[i];
        WqT[o] = (_Float16)Wq[i];
        WvT[o] = (_Float16)Wv[i];
    }
}

// ------------------------------------------------------------------ proj ----
// One block = 16 rows of x (one WMMA M-tile), 4 waves = 4 slices of H.
__global__ __launch_bounds__(128) void proj_kernel(
    const float*    __restrict__ x,
    const _Float16* __restrict__ WqT,
    const _Float16* __restrict__ WkT,
    const _Float16* __restrict__ WvT,
    _Float16* __restrict__ Qh,      // [B*T][H], pre-scaled by H^-0.5
    _Float16* __restrict__ Kh,      // [B*T][H]
    _Float16* __restrict__ Vt)      // [B][H][T]
{
    __shared__ __align__(32) _Float16 xs[16 * 32];      // A tile, f16

    const int tile  = blockIdx.x;                       // 0..B*T/16-1
    const int row0  = tile * 16;                        // flat row into [B*T]
    const int tid   = threadIdx.x;
    const int wave  = tid >> 5;
    const int lane  = tid & 31;
    const int m     = lane & 15;                        // row (A) / col (B,C)
    const int hig   = lane >> 4;
    const int koff8 = hig * 8;                          // unused; A-frag K base is kb16
    const int kb16  = hig * 16;                         // B-frag K base
    const int n0    = wave * 16;                        // H slice

    v8f cq = {}; v8f ck = {}; v8f cv = {};

    for (int kc = 0; kc < Cd; kc += 32) {
        __syncthreads();
        // cooperative: x[row0..row0+15][kc..kc+31] f32 -> f16 LDS tile
        {
            int r  = tid >> 3;                          // 0..15
            int c4 = (tid & 7) * 4;                     // 0..28
            const float4 xv = *(const float4*)(x + (size_t)(row0 + r) * Cd + kc + c4);
            _Float16* d = &xs[r * 32 + c4];
            d[0] = (_Float16)xv.x; d[1] = (_Float16)xv.y;
            d[2] = (_Float16)xv.z; d[3] = (_Float16)xv.w;
        }
        __syncthreads();

        // A fragment (16x32 f16): lo lanes K 0..15, hi lanes 16..31
        v16h a;
        {
            v8h lo  = *(const v8h*)&xs[m * 32 + kb16];
            v8h hi8 = *(const v8h*)&xs[m * 32 + kb16 + 8];
#pragma unroll
            for (int j = 0; j < 8; ++j) { a[j] = lo[j]; a[8 + j] = hi8[j]; }
        }

        // B fragments: lane n holds col h=n0+n, 16 consecutive K (contiguous 32B)
        const size_t wb = (size_t)(n0 + m) * Cd + kc + kb16;
        v16h bq = *(const v16h*)(WqT + wb);
        v16h bk = *(const v16h*)(WkT + wb);
        v16h bv = *(const v16h*)(WvT + wb);

        cq = WMMA_F16(a, bq, cq);
        ck = WMMA_F16(a, bk, ck);
        cv = WMMA_F16(a, bv, cv);
    }

    // Epilogue. C-frag element r = (row m'=r+8*hig, col n=m).
    const int b  = row0 / Td;
    const int t0 = row0 % Td;
#pragma unroll
    for (int r = 0; r < 8; ++r) {
        int mm = r + 8 * hig;
        size_t idx = (size_t)(row0 + mm) * Hd + n0 + m;
        Qh[idx] = (_Float16)(cq[r] * 0.125f);           // fold H^-0.5 = 1/8
        Kh[idx] = (_Float16)ck[r];
    }
    {   // V transposed: lane's 8 values are consecutive t -> one b128 store
        v8h pv;
#pragma unroll
        for (int r = 0; r < 8; ++r) pv[r] = (_Float16)cv[r];
        size_t vidx = ((size_t)b * Hd + n0 + m) * Td + t0 + 8 * hig;
        *(v8h*)(Vt + vidx) = pv;
    }
}

// ------------------------------------------------------------------ attn ----
// One block = 4 waves = 4 consecutive 16-query tiles of the same batch.
// K/V 32-key blocks are staged once per workgroup in double-buffered LDS.
__global__ __launch_bounds__(128) void attn_kernel(
    const _Float16* __restrict__ Qh,
    const _Float16* __restrict__ Kh,
    const _Float16* __restrict__ Vt,
    float* __restrict__ out)
{
    __shared__ __align__(32) _Float16 ks[2][32 * 64];   // K tile [key][h], 2x4KB
    __shared__ __align__(32) _Float16 vs[2][64 * 32];   // V tile [h][key], 2x4KB
    __shared__ __align__(32) _Float16 ps[4][16 * 32];   // per-wave P tile

    const int tid   = threadIdx.x;
    const int wave  = tid >> 5;
    const int lane  = tid & 31;
    const int m     = lane & 15;
    const int hig   = lane >> 4;
    const int koff8 = hig * 8;
    const int kb16  = hig * 16;

    const int tileId = blockIdx.x * 4 + wave;           // 0..B*T/16-1
    const int b  = tileId >> 7;                         // 128 tiles per batch
    const int q0 = (tileId & 127) * 16;
    // block-uniform trip count: largest tile in this block (never crosses batch)
    const int q0max = ((blockIdx.x * 4 + 3) & 127) * 16;
    const int nblk  = (q0max + 16 + 31) >> 5;

    // cooperative K/V staging map: 32B (v16h) per thread per tile
    const int kkey = tid >> 2;                          // 0..31
    const int khof = (tid & 3) * 16;                    // 0/16/32/48
    const int vh   = tid >> 1;                          // 0..63
    const int vkof = (tid & 1) * 16;                    // 0/16
    const _Float16* kgb = Kh + (size_t)b * Td * Hd;     // + (k0+kkey)*H + khof
    const _Float16* vgb = Vt + (size_t)b * Hd * Td;     // + vh*T + k0 + vkof

    // ---- Q A-fragments for K(h)=0..31 and 32..63 (scale pre-folded)
    const _Float16* qbase = Qh + ((size_t)b * Td + q0) * Hd + m * Hd;
    v16h aq0, aq1;
    {
        v8h a0 = *(const v8h*)(qbase + kb16);
        v8h a1 = *(const v8h*)(qbase + kb16 + 8);
        v8h a2 = *(const v8h*)(qbase + 32 + kb16);
        v8h a3 = *(const v8h*)(qbase + 40 + kb16);
#pragma unroll
        for (int j = 0; j < 8; ++j) {
            aq0[j] = a0[j]; aq0[8 + j] = a1[j];
            aq1[j] = a2[j]; aq1[8 + j] = a3[j];
        }
    }

    v8f o0 = {}, o1 = {}, o2 = {}, o3 = {};
    float mi[8], li[8];
#pragma unroll
    for (int r = 0; r < 8; ++r) { mi[r] = -1e30f; li[r] = 0.0f; }

    // ---- stage block 0
    v16h kreg = *(const v16h*)(kgb + (size_t)kkey * Hd + khof);
    v16h vreg = *(const v16h*)(vgb + (size_t)vh * Td + vkof);
    *(v16h*)&ks[0][kkey * 64 + khof] = kreg;
    *(v16h*)&vs[0][vh * 32 + vkof]   = vreg;
    __syncthreads();

    for (int kb = 0; kb < nblk; ++kb) {
        const int k0  = kb * 32;
        const int cur = kb & 1;
        const int nxt = cur ^ 1;

        // issue global prefetch of next K/V block (consumed at loop bottom)
        if (kb + 1 < nblk) {
            kreg = *(const v16h*)(kgb + (size_t)(k0 + 32 + kkey) * Hd + khof);
            vreg = *(const v16h*)(vgb + (size_t)vh * Td + k0 + 32 + vkof);
        }

        // ---- S = Q K^T : two 16x16 f32 tiles from the shared K tile
        v8f s0 = {}, s1 = {};
        {
            v16h b0c0 = *(const v16h*)&ks[cur][m * 64 + kb16];
            v16h b0c1 = *(const v16h*)&ks[cur][m * 64 + 32 + kb16];
            v16h b1c0 = *(const v16h*)&ks[cur][(16 + m) * 64 + kb16];
            v16h b1c1 = *(const v16h*)&ks[cur][(16 + m) * 64 + 32 + kb16];
            s0 = WMMA_F16(aq0, b0c0, s0);
            s0 = WMMA_F16(aq1, b0c1, s0);
            s1 = WMMA_F16(aq0, b1c0, s1);
            s1 = WMMA_F16(aq1, b1c1, s1);
        }

        // ---- causal mask (diagonal-touching and beyond-diagonal blocks)
        if (k0 + 31 >= q0) {
#pragma unroll
            for (int r = 0; r < 8; ++r) {
                int qi = q0 + r + 8 * hig;
                if (k0 + m > qi)      s0[r] = -1e30f;
                if (k0 + 16 + m > qi) s1[r] = -1e30f;
            }
        }

        // ---- online softmax (row stats across the 16-lane half-groups)
        float alpha[8];
#pragma unroll
        for (int r = 0; r < 8; ++r) {
            float mx = fmaxf(s0[r], s1[r]);
#pragma unroll
            for (int off = 8; off >= 1; off >>= 1)
                mx = fmaxf(mx, __shfl_xor(mx, off, 32));
            float mnew = fmaxf(mi[r], mx);
            float a  = __expf(mi[r] - mnew);
            float p0 = __expf(s0[r] - mnew);
            float p1 = __expf(s1[r] - mnew);
            float rs = p0 + p1;
#pragma unroll
            for (int off = 8; off >= 1; off >>= 1)
                rs += __shfl_xor(rs, off, 32);
            li[r] = li[r] * a + rs;
            mi[r] = mnew;
            alpha[r] = a;
            int mm = r + 8 * hig;                       // C-layout -> LDS P tile
            ps[wave][mm * 32 + m]      = (_Float16)p0;
            ps[wave][mm * 32 + 16 + m] = (_Float16)p1;
        }
#pragma unroll
        for (int r = 0; r < 8; ++r) {
            o0[r] *= alpha[r]; o1[r] *= alpha[r];
            o2[r] *= alpha[r]; o3[r] *= alpha[r];
        }

        // ---- P back as A-fragment (same-wave LDS, DS ops are in-order)
        v16h pa;
        {
            v8h lo  = *(const v8h*)&ps[wave][m * 32 + kb16];
            v8h hi8 = *(const v8h*)&ps[wave][m * 32 + kb16 + 8];
#pragma unroll
            for (int j = 0; j < 8; ++j) { pa[j] = lo[j]; pa[8 + j] = hi8[j]; }
        }

        // ---- O += P V from the shared V tile (h-major, keys contiguous)
        {
            v16h bv0 = *(const v16h*)&vs[cur][(0 * 16 + m) * 32 + kb16];
            v16h bv1 = *(const v16h*)&vs[cur][(1 * 16 + m) * 32 + kb16];
            v16h bv2 = *(const v16h*)&vs[cur][(2 * 16 + m) * 32 + kb16];
            v16h bv3 = *(const v16h*)&vs[cur][(3 * 16 + m) * 32 + kb16];
            o0 = WMMA_F16(pa, bv0, o0);
            o1 = WMMA_F16(pa, bv1, o1);
            o2 = WMMA_F16(pa, bv2, o2);
            o3 = WMMA_F16(pa, bv3, o3);
        }

        // ---- publish prefetched block into the other buffer
        if (kb + 1 < nblk) {
            *(v16h*)&ks[nxt][kkey * 64 + khof] = kreg;
            *(v16h*)&vs[nxt][vh * 32 + vkof]   = vreg;
        }
        __syncthreads();
    }

    // ---- normalize and store fp32 out[b][t][h]
    float* ob = out + ((size_t)b * Td + q0) * Hd;
#pragma unroll
    for (int r = 0; r < 8; ++r) {
        float inv = 1.0f / li[r];
        int mm = r + 8 * hig;
        ob[(size_t)mm * Hd +  0 + m] = o0[r] * inv;
        ob[(size_t)mm * Hd + 16 + m] = o1[r] * inv;
        ob[(size_t)mm * Hd + 32 + m] = o2[r] * inv;
        ob[(size_t)mm * Hd + 48 + m] = o3[r] * inv;
    }
}

// ---------------------------------------------------------------- launch ----
extern "C" void kernel_launch(void* const* d_in, const int* in_sizes, int n_in,
                              void* d_out, int out_size, void* d_ws, size_t ws_size,
                              hipStream_t stream) {
    const float* x  = (const float*)d_in[0];
    const float* Wk = (const float*)d_in[1];
    const float* Wq = (const float*)d_in[2];
    const float* Wv = (const float*)d_in[3];
    float* out = (float*)d_out;

    char* ws = (char*)d_ws;
    const size_t wT  = (size_t)Hd * Cd * sizeof(_Float16);      // 128 KB
    const size_t qkv = (size_t)Bd * Td * Hd * sizeof(_Float16); // 2 MB
    _Float16* WkT = (_Float16*)(ws + 0 * wT);
    _Float16* WqT = (_Float16*)(ws + 1 * wT);
    _Float16* WvT = (_Float16*)(ws + 2 * wT);
    _Float16* Qh  = (_Float16*)(ws + 3 * wT);
    _Float16* Kh  = (_Float16*)(ws + 3 * wT + qkv);
    _Float16* Vt  = (_Float16*)(ws + 3 * wT + 2 * qkv);

    prep_w_kernel<<<(Cd * Hd + 255) / 256, 256, 0, stream>>>(Wk, Wq, Wv, WkT, WqT, WvT);
    proj_kernel<<<Bd * Td / 16, 128, 0, stream>>>(x, WqT, WkT, WvT, Qh, Kh, Vt);
    attn_kernel<<<Bd * Td / 16 / 4, 128, 0, stream>>>(Qh, Kh, Vt, out);
}
